// MaskedAutoencoderViT_39221641347753
// MI455X (gfx1250) — matrix operand order repaired and checked by reading, not compiled, not run on hardware
//
#include <hip/hip_runtime.h>
#include <hip/hip_bf16.h>
#include <math.h>
#include <stdint.h>

typedef _Float16 h16v __attribute__((ext_vector_type(16)));
typedef _Float16 h8v  __attribute__((ext_vector_type(8)));
typedef float    f8v  __attribute__((ext_vector_type(8)));
typedef int      i32x4 __attribute__((vector_size(16)));   // matches builtin param

namespace {
constexpr int   Dn   = 2;
constexpr int   Cc   = 1024;
constexpr int   NHd  = 16;
constexpr int   KDd  = 64;
constexpr int   HSd  = 14;
constexpr int   Np   = HSd * HSd;     // 196
constexpr int   HID  = 4096;
constexpr int   Bb   = 32;
constexpr int   MTOK = Bb * Np;       // 6272 = 49 * 128
constexpr float KSCALE = 0.125f;      // 64^-0.5
constexpr float LNEPS  = 1e-5f;
}

// --- CDNA5 async global->LDS copy (ASYNCcnt-tracked), with sync fallback ----
#if __has_builtin(__builtin_amdgcn_global_load_async_to_lds_b128)
#define HAVE_ASYNC_LDS 1
#else
#define HAVE_ASYNC_LDS 0
#endif

#if __has_builtin(__builtin_amdgcn_s_wait_asynccnt)
#define WAIT_ASYNC(n) __builtin_amdgcn_s_wait_asynccnt(n)
#else
#define WAIT_ASYNC(n) asm volatile("s_wait_asynccnt " #n ::: "memory")
#endif

typedef __attribute__((address_space(1))) i32x4* gvec_p;   // global, 64-bit
typedef __attribute__((address_space(3))) i32x4* lvec_p;   // LDS, 32-bit

__device__ __forceinline__ void stage16B(const _Float16* g, _Float16* l) {
#if HAVE_ASYNC_LDS
  __builtin_amdgcn_global_load_async_to_lds_b128(
      (gvec_p)(uintptr_t)g,
      (lvec_p)(uint32_t)(uintptr_t)l,   // flat->LDS: addr[31:0] per ISA
      0, 0);
#else
  *(h8v*)l = *(const h8v*)g;
#endif
}

__device__ __forceinline__ h16v cat16(h8v lo, h8v hi) {
  return __builtin_shufflevector(lo, hi, 0, 1, 2, 3, 4, 5, 6, 7,
                                 8, 9, 10, 11, 12, 13, 14, 15);
}

// ---------------------------------------------------------------------------
// Elementwise helpers
// ---------------------------------------------------------------------------
__global__ void copy_f32_kernel(const float* __restrict__ src,
                                float* __restrict__ dst, int n) {
  int i = blockIdx.x * blockDim.x + threadIdx.x;
  if (i < n) dst[i] = src[i];
}

__global__ void cvt_f32_f16_kernel(const float* __restrict__ src,
                                   _Float16* __restrict__ dst, int n) {
  int i = blockIdx.x * blockDim.x + threadIdx.x;
  if (i < n) dst[i] = (_Float16)src[i];
}

__global__ void add_cvt_f16_kernel(const float* __restrict__ a,
                                   const float* __restrict__ b,
                                   _Float16* __restrict__ dst, int n) {
  int i = blockIdx.x * blockDim.x + threadIdx.x;
  if (i < n) dst[i] = (_Float16)(a[i] + b[i]);
}

// ---------------------------------------------------------------------------
// LayerNorm over C=1024 per patch token -> f16 activations (row-major M x C)
// ---------------------------------------------------------------------------
__global__ __launch_bounds__(256)
void ln_kernel(const float* __restrict__ tok,
               const float* __restrict__ gamma,
               const float* __restrict__ beta,
               _Float16* __restrict__ dst) {
  const int m    = blockIdx.x;          // 0..MTOK-1
  const int bidx = m / Np;
  const int tix  = m % Np;
  const float* x = tok + ((size_t)bidx * (Np + 1) + 1 + tix) * Cc;
  __shared__ float red[256];
  const int tid = threadIdx.x;

  float v[4];
  float s = 0.f;
#pragma unroll
  for (int i = 0; i < 4; ++i) { v[i] = x[tid + 256 * i]; s += v[i]; }
  red[tid] = s; __syncthreads();
  for (int off = 128; off > 0; off >>= 1) {
    if (tid < off) red[tid] += red[tid + off];
    __syncthreads();
  }
  const float mean = red[0] * (1.f / Cc);
  __syncthreads();

  float q = 0.f;
#pragma unroll
  for (int i = 0; i < 4; ++i) { float d = v[i] - mean; q += d * d; }
  red[tid] = q; __syncthreads();
  for (int off = 128; off > 0; off >>= 1) {
    if (tid < off) red[tid] += red[tid + off];
    __syncthreads();
  }
  const float rstd = rsqrtf(red[0] * (1.f / Cc) + LNEPS);

#pragma unroll
  for (int i = 0; i < 4; ++i) {
    int c = tid + 256 * i;
    dst[(size_t)m * Cc + c] = (_Float16)((v[i] - mean) * rstd * gamma[c] + beta[c]);
  }
}

// ---------------------------------------------------------------------------
// WMMA GEMM: out[M,N] = A[M,K] (f16, row-major) x W[N,K]^T (f16, row-major) + bias
//
// Block = 8 waves = M-tile 128 x N-tile 64.  The 64x32 B slices are shared by
// all waves, staged in LDS double-buffered via GLOBAL_LOAD_ASYNC_TO_LDS_B128
// (ASYNCcnt).  K loop steps 64 with two FIXED register/buffer slots so there
// is no register rotation (no WMMA->VALU hazard NOPs) and only one branch per
// 64-K.  LDS B rows padded to 40 halves -> conflict-free ds_load_b128 reads.
//
// MODE 0: out f32 = (acc + bias[n]) * scale
// MODE 1: tok[b, 1+t, n] += acc + bias[n]    (residual, skips cls token)
// MODE 2: out16 f16 = tanh(alpha*(acc+bias))*dyt_w[n]+dyt_b[n]   (DyT epilogue)
// ---------------------------------------------------------------------------
template <int MODE>
__global__ __launch_bounds__(256)
void gemm_wmma_kernel(const _Float16* __restrict__ A,
                      const _Float16* __restrict__ W,
                      const float* __restrict__ bias,
                      float* __restrict__ out,
                      int M, int N, int K, float scale,
                      float* __restrict__ tok,
                      const float* __restrict__ alpha,
                      const float* __restrict__ dyt_w,
                      const float* __restrict__ dyt_b,
                      _Float16* __restrict__ out16) {
  constexpr int BSTR = 40;                       // padded halves per B row
  __shared__ _Float16 bsm[2][64 * BSTR];         // 10 KB

  const int tid  = threadIdx.x;
  const int wid  = tid >> 5;
  const int lane = tid & 31;
  const int ng   = N >> 6;                       // 64-wide n groups
  const int mt8  = blockIdx.x / ng;              // 128-row group
  const int ngrp = blockIdx.x % ng;
  const int m0   = (mt8 * 8 + wid) << 4;
  const int n0   = ngrp << 6;

  const int  lr = lane & 15;
  const bool hi = lane >= 16;
  const int  bko = hi ? 8 : 0;                   // B fragment K sub-offset

  // staging assignment: thread copies one 16B chunk of the 64x32 B slice
  const int srow = tid >> 2;                     // 0..63
  const int schk = tid & 3;                      // which 8-half chunk
  const _Float16* gsrc = W + (size_t)(n0 + srow) * K + schk * 8;
  _Float16* ldst0 = &bsm[0][srow * BSTR + schk * 8];
  _Float16* ldst1 = &bsm[1][srow * BSTR + schk * 8];

  // A fragment source: row m0+lr, halves {kb..kb+7, kb+16..kb+23}, kb = hi?8:0
  const _Float16* arow = A + (size_t)(m0 + lr) * K + (hi ? 8 : 0);

  f8v acc[4] = {};

  auto compute = [&](int buf, h16v a) {
#pragma unroll
    for (int t = 0; t < 4; ++t) {
      const _Float16* bp = &bsm[buf][(16 * t + lr) * BSTR + bko];
      h16v b = cat16(*(const h8v*)bp, *(const h8v*)(bp + 16));
      acc[t] = __builtin_amdgcn_wmma_f32_16x16x32_f16(
          /*neg_a=*/false, a, /*neg_b=*/false, b,
          /*c_mod=*/(short)0, acc[t], /*reuse_a=*/false, /*reuse_b=*/false);
    }
  };

  // prologue: slice0 (kk=0) -> buf0, A regs slot0
  stage16B(gsrc, ldst0);
  h8v alo0 = *(const h8v*)(arow);
  h8v ahi0 = *(const h8v*)(arow + 16);

  for (int kk = 0; kk < K; kk += 64) {
    // stage slice1 (kk+32) -> buf1, prefetch A slot1
    stage16B(gsrc + kk + 32, ldst1);
    h8v alo1 = *(const h8v*)(arow + kk + 32);
    h8v ahi1 = *(const h8v*)(arow + kk + 48);
#if HAVE_ASYNC_LDS
    WAIT_ASYNC(1);                               // slice0 landed; slice1 in flight
#endif
    __syncthreads();
    compute(0, cat16(alo0, ahi0));
    __syncthreads();                             // buf0 reads done

    if (kk + 64 < K) {
      // stage next slice0 (kk+64) -> buf0, prefetch A slot0
      stage16B(gsrc + kk + 64, ldst0);
      alo0 = *(const h8v*)(arow + kk + 64);
      ahi0 = *(const h8v*)(arow + kk + 80);
#if HAVE_ASYNC_LDS
      WAIT_ASYNC(1);                             // slice1 landed; next in flight
#endif
    } else {
#if HAVE_ASYNC_LDS
      WAIT_ASYNC(0);                             // drain final slice
#endif
    }
    __syncthreads();
    compute(1, cat16(alo1, ahi1));
    __syncthreads();                             // buf1 reads done
  }

  // Epilogue. C/D layout: VGPR j -> M = m0 + j + (hi?8:0); N = n0 + 16t + lr.
  const int mbase = m0 + (hi ? 8 : 0);
#pragma unroll
  for (int t = 0; t < 4; ++t) {
    const int n  = n0 + 16 * t + lr;
    const float bn = bias[n];
#pragma unroll
    for (int j = 0; j < 8; ++j) {
      const int m = mbase + j;
      float y = (acc[t][j] + bn) * scale;
      if (MODE == 0) {
        out[(size_t)m * N + n] = y;
      } else if (MODE == 1) {
        const int bidx = m / Np;
        const int tix  = m % Np;
        tok[((size_t)bidx * (Np + 1) + 1 + tix) * Cc + n] += y;
      } else {
        float z = tanhf(alpha[0] * y) * dyt_w[n] + dyt_b[n];
        out16[(size_t)m * N + n] = (_Float16)z;
      }
    }
  }
}

// ---------------------------------------------------------------------------
// 5x5 depthwise conv (LePE), padding 2, NHWC, one thread per output element.
// ---------------------------------------------------------------------------
__global__ __launch_bounds__(256)
void lepe_kernel(const float* __restrict__ v,
                 const float* __restrict__ w,
                 const float* __restrict__ bias,
                 float* __restrict__ out) {
  int idx = blockIdx.x * blockDim.x + threadIdx.x;
  if (idx >= MTOK * Cc) return;
  const int c  = idx & (Cc - 1);
  int t        = idx >> 10;
  const int wq = t % HSd; t /= HSd;
  const int hq = t % HSd;
  const int bq = t / HSd;
  float acc = bias[c];
#pragma unroll
  for (int kh = 0; kh < 5; ++kh) {
    const int hh = hq + kh - 2;
    if (hh < 0 || hh >= HSd) continue;
#pragma unroll
    for (int kw = 0; kw < 5; ++kw) {
      const int ww = wq + kw - 2;
      if (ww < 0 || ww >= HSd) continue;
      acc += v[((size_t)bq * Np + hh * HSd + ww) * Cc + c] * w[(kh * 5 + kw) * Cc + c];
    }
  }
  out[idx] = acc;
}

// ---------------------------------------------------------------------------
// Rotary theta-shift, in place on (MTOK, C) f32.  One thread per channel pair.
// ---------------------------------------------------------------------------
__global__ __launch_bounds__(256)
void theta_kernel(float* __restrict__ x) {
  int idx = blockIdx.x * blockDim.x + threadIdx.x;
  if (idx >= MTOK * (Cc / 2)) return;
  const int pair = idx & (Cc / 2 - 1);   // head*32 + p
  const int m    = idx >> 9;
  const int nidx = m % Np;               // spatial index h*14+w
  const int p    = pair & 31;            // pair within head (half = 32)
  const float ang = (float)nidx * powf(10000.f, -(float)p * (1.f / 31.f));
  const float s = sinf(ang);
  const float c = cosf(ang);
  float* px = x + (size_t)m * Cc + 2 * pair;
  const float x0 = px[0], x1 = px[1];
  px[0] = x0 * c - x1 * s;
  px[1] = x1 * c + x0 * s;
}

// ---------------------------------------------------------------------------
// Axial attention (14-wide softmax attention), one block per (b, head, fix).
// axis 0: fix = row h, attend over width.  axis 1: fix = col w, attend over rows.
// ---------------------------------------------------------------------------
__global__ __launch_bounds__(256)
void attn_axis_kernel(const float* __restrict__ qr,
                      const float* __restrict__ kr,
                      const float* __restrict__ vin,
                      float* __restrict__ out, int axis) {
  const int blk  = blockIdx.x;          // B*NH*HS
  const int fix  = blk % HSd;
  const int tmp  = blk / HSd;
  const int head = tmp % NHd;
  const int b    = tmp / NHd;

  __shared__ float qs[HSd][KDd + 1];
  __shared__ float ks[HSd][KDd + 1];
  __shared__ float vs[HSd][KDd + 1];
  __shared__ float ss[HSd][HSd + 2];

  const float decay = logf(1.f - exp2f(-1.f - (float)head));

  for (int i = threadIdx.x; i < HSd * KDd; i += 256) {
    const int u = i >> 6, d = i & 63;
    const int t = (axis == 0) ? (fix * HSd + u) : (u * HSd + fix);
    const size_t off = ((size_t)b * Np + t) * Cc + head * KDd + d;
    qs[u][d] = qr[off];
    ks[u][d] = kr[off];
    vs[u][d] = vin[off];
  }
  __syncthreads();

  for (int i = threadIdx.x; i < HSd * HSd; i += 256) {
    const int w = i / HSd, u = i % HSd;
    float s = 0.f;
#pragma unroll
    for (int d = 0; d < KDd; ++d) s += qs[w][d] * ks[u][d];
    ss[w][u] = s + fabsf((float)(w - u)) * decay;
  }
  __syncthreads();

  if (threadIdx.x < HSd) {
    const int w = threadIdx.x;
    float mx = -1e30f;
    for (int u = 0; u < HSd; ++u) mx = fmaxf(mx, ss[w][u]);
    float sum = 0.f;
    for (int u = 0; u < HSd; ++u) { float e = expf(ss[w][u] - mx); ss[w][u] = e; sum += e; }
    const float inv = 1.f / sum;
    for (int u = 0; u < HSd; ++u) ss[w][u] *= inv;
  }
  __syncthreads();

  for (int i = threadIdx.x; i < HSd * KDd; i += 256) {
    const int w = i >> 6, d = i & 63;
    float s = 0.f;
#pragma unroll
    for (int u = 0; u < HSd; ++u) s += ss[w][u] * vs[u][d];
    const int t = (axis == 0) ? (fix * HSd + w) : (w * HSd + fix);
    out[((size_t)b * Np + t) * Cc + head * KDd + d] = s;
  }
}

// ---------------------------------------------------------------------------
// Host-side launch
// ---------------------------------------------------------------------------
extern "C" void kernel_launch(void* const* d_in, const int* in_sizes, int n_in,
                              void* d_out, int out_size, void* d_ws, size_t ws_size,
                              hipStream_t stream) {
  const float* x      = (const float*)d_in[0];
  const float* ln1_w  = (const float*)d_in[1];
  const float* ln1_b  = (const float*)d_in[2];
  const float* q_w    = (const float*)d_in[3];
  const float* q_b    = (const float*)d_in[4];
  const float* k_w    = (const float*)d_in[5];
  const float* k_b    = (const float*)d_in[6];
  const float* v_w    = (const float*)d_in[7];
  const float* v_b    = (const float*)d_in[8];
  const float* lepe_w = (const float*)d_in[9];
  const float* lepe_b = (const float*)d_in[10];
  const float* out_w  = (const float*)d_in[11];
  const float* out_b  = (const float*)d_in[12];
  const float* ln2_w  = (const float*)d_in[13];
  const float* ln2_b  = (const float*)d_in[14];
  const float* fc1_w  = (const float*)d_in[15];
  const float* fc1_b  = (const float*)d_in[16];
  const float* alpha  = (const float*)d_in[17];
  const float* dyt_w  = (const float*)d_in[18];
  const float* dyt_b  = (const float*)d_in[19];
  const float* fc2_w  = (const float*)d_in[20];
  const float* fc2_b  = (const float*)d_in[21];

  char* ws = (char*)d_ws;
  size_t off = 0;
  auto alloc = [&](size_t bytes) -> void* {
    off = (off + 255) & ~(size_t)255;
    void* p = ws + off;
    off += bytes;
    return p;
  };

  float*    tok    = (float*)alloc((size_t)Bb * (Np + 1) * Cc * 4);
  _Float16* act16  = (_Float16*)alloc((size_t)MTOK * Cc * 2);
  _Float16* h16    = (_Float16*)alloc((size_t)MTOK * HID * 2);
  float*    qbuf   = (float*)alloc((size_t)MTOK * Cc * 4);
  float*    kbuf   = (float*)alloc((size_t)MTOK * Cc * 4);
  float*    vbuf   = (float*)alloc((size_t)MTOK * Cc * 4);
  float*    lepe   = (float*)alloc((size_t)MTOK * Cc * 4);
  float*    attnW  = (float*)alloc((size_t)MTOK * Cc * 4);
  float*    attnO  = (float*)alloc((size_t)MTOK * Cc * 4);
  _Float16* qw16   = (_Float16*)alloc((size_t)Dn * Cc * Cc * 2);
  _Float16* kw16   = (_Float16*)alloc((size_t)Dn * Cc * Cc * 2);
  _Float16* vw16   = (_Float16*)alloc((size_t)Dn * Cc * Cc * 2);
  _Float16* ow16   = (_Float16*)alloc((size_t)Dn * Cc * Cc * 2);
  _Float16* f1w16  = (_Float16*)alloc((size_t)Dn * HID * Cc * 2);
  _Float16* f2w16  = (_Float16*)alloc((size_t)Dn * Cc * HID * 2);

  auto cdiv = [](int a, int b) { return (a + b - 1) / b; };

  // One-time weight conversions (covers both layers).
  {
    int n1 = Dn * Cc * Cc;
    int n2 = Dn * HID * Cc;
    cvt_f32_f16_kernel<<<cdiv(n1, 256), 256, 0, stream>>>(q_w,   qw16,  n1);
    cvt_f32_f16_kernel<<<cdiv(n1, 256), 256, 0, stream>>>(k_w,   kw16,  n1);
    cvt_f32_f16_kernel<<<cdiv(n1, 256), 256, 0, stream>>>(v_w,   vw16,  n1);
    cvt_f32_f16_kernel<<<cdiv(n1, 256), 256, 0, stream>>>(out_w, ow16,  n1);
    cvt_f32_f16_kernel<<<cdiv(n2, 256), 256, 0, stream>>>(fc1_w, f1w16, n2);
    cvt_f32_f16_kernel<<<cdiv(n2, 256), 256, 0, stream>>>(fc2_w, f2w16, n2);
  }

  // tok = x
  {
    int n = Bb * (Np + 1) * Cc;
    copy_f32_kernel<<<cdiv(n, 256), 256, 0, stream>>>(x, tok, n);
  }

  const int nElem  = MTOK * Cc;
  const int gElem  = cdiv(nElem, 256);
  const int gPairs = cdiv(MTOK * (Cc / 2), 256);
  const int gAttn  = Bb * NHd * HSd;
  auto gemm_grid = [](int M, int N) { return (M / 128) * (N / 64); };

  for (int d = 0; d < Dn; ++d) {
    const _Float16* qw = qw16  + (size_t)d * Cc * Cc;
    const _Float16* kw = kw16  + (size_t)d * Cc * Cc;
    const _Float16* vw = vw16  + (size_t)d * Cc * Cc;
    const _Float16* ow = ow16  + (size_t)d * Cc * Cc;
    const _Float16* f1 = f1w16 + (size_t)d * HID * Cc;
    const _Float16* f2 = f2w16 + (size_t)d * Cc * HID;

    // 1) LN1 -> f16 activations
    ln_kernel<<<MTOK, 256, 0, stream>>>(tok, ln1_w + d * Cc, ln1_b + d * Cc, act16);

    // 2) q, k (scaled), v projections (WMMA)
    gemm_wmma_kernel<0><<<gemm_grid(MTOK, Cc), 256, 0, stream>>>(
        act16, qw, q_b + d * Cc, qbuf, MTOK, Cc, Cc, 1.f,
        nullptr, nullptr, nullptr, nullptr, nullptr);
    gemm_wmma_kernel<0><<<gemm_grid(MTOK, Cc), 256, 0, stream>>>(
        act16, kw, k_b + d * Cc, kbuf, MTOK, Cc, Cc, KSCALE,
        nullptr, nullptr, nullptr, nullptr, nullptr);
    gemm_wmma_kernel<0><<<gemm_grid(MTOK, Cc), 256, 0, stream>>>(
        act16, vw, v_b + d * Cc, vbuf, MTOK, Cc, Cc, 1.f,
        nullptr, nullptr, nullptr, nullptr, nullptr);

    // 3) LePE depthwise conv on v
    lepe_kernel<<<gElem, 256, 0, stream>>>(vbuf, lepe_w + (size_t)d * 25 * Cc,
                                           lepe_b + d * Cc, lepe);

    // 4) rotary theta-shift on q and k (in place)
    theta_kernel<<<gPairs, 256, 0, stream>>>(qbuf);
    theta_kernel<<<gPairs, 256, 0, stream>>>(kbuf);

    // 5) axial attention: width axis then height axis
    attn_axis_kernel<<<gAttn, 256, 0, stream>>>(qbuf, kbuf, vbuf, attnW, 0);
    attn_axis_kernel<<<gAttn, 256, 0, stream>>>(qbuf, kbuf, attnW, attnO, 1);

    // 6) (attn + lepe) -> f16, then out projection with residual into tok
    add_cvt_f16_kernel<<<gElem, 256, 0, stream>>>(attnO, lepe, act16, nElem);
    gemm_wmma_kernel<1><<<gemm_grid(MTOK, Cc), 256, 0, stream>>>(
        act16, ow, out_b + d * Cc, nullptr, MTOK, Cc, Cc, 1.f,
        tok, nullptr, nullptr, nullptr, nullptr);

    // 7) LN2 -> f16, fc1 with fused DyT epilogue -> f16 hidden
    ln_kernel<<<MTOK, 256, 0, stream>>>(tok, ln2_w + d * Cc, ln2_b + d * Cc, act16);
    gemm_wmma_kernel<2><<<gemm_grid(MTOK, HID), 256, 0, stream>>>(
        act16, f1, fc1_b + d * HID, nullptr, MTOK, HID, Cc, 1.f,
        nullptr, alpha + d, dyt_w + d * HID, dyt_b + d * HID, h16);

    // 8) fc2 with residual into tok
    gemm_wmma_kernel<1><<<gemm_grid(MTOK, Cc), 256, 0, stream>>>(
        h16, f2, fc2_b + d * Cc, nullptr, MTOK, Cc, HID, 1.f,
        tok, nullptr, nullptr, nullptr, nullptr);
  }

  // tok -> d_out
  {
    int n = Bb * (Np + 1) * Cc;
    copy_f32_kernel<<<cdiv(n, 256), 256, 0, stream>>>(tok, (float*)d_out, n);
  }
}
